// RouterCNN_26998164423125
// MI455X (gfx1250) — matrix operand order repaired and checked by reading, not compiled, not run on hardware
//
#include <hip/hip_runtime.h>
#include <hip/hip_bf16.h>

typedef _Float16 v16h __attribute__((ext_vector_type(16)));
typedef _Float16 v8h  __attribute__((ext_vector_type(8)));
typedef float    v8f  __attribute__((ext_vector_type(8)));

#define BTOK 8192
#define DM   4096

// ---------------------------------------------------------------------------
// trans_w [K][N] fp32  ->  Wt [N][K] f16   (tiled transpose + convert)
// ---------------------------------------------------------------------------
__global__ __launch_bounds__(256)
void transpose_convert_kernel(const float* __restrict__ W, _Float16* __restrict__ Wt) {
    __shared__ float tile[32][33];
    const int tx = threadIdx.x & 31;
    const int ty = threadIdx.x >> 5;           // 0..7
    const int nb = blockIdx.x * 32;
    const int kb = blockIdx.y * 32;
#pragma unroll
    for (int i = 0; i < 4; ++i) {
        const int r = ty + i * 8;
        tile[r][tx] = W[(size_t)(kb + r) * DM + nb + tx];
    }
    __syncthreads();
#pragma unroll
    for (int i = 0; i < 4; ++i) {
        const int r = ty + i * 8;
        Wt[(size_t)(nb + r) * DM + kb + tx] = (_Float16)tile[tx][r];
    }
}

// ---------------------------------------------------------------------------
// conv1: x[B,3,32,32] --3x3 s2 p1--> relu -> flat [B, 4096] fp32
// ---------------------------------------------------------------------------
__global__ __launch_bounds__(256)
void conv1_kernel(const float* __restrict__ x, const float* __restrict__ w,
                  const float* __restrict__ bsrc, float* __restrict__ out) {
    __shared__ float xs[3 * 34 * 34];
    __shared__ float ws[16 * 27];
    __shared__ float bb[16];
    const int tid = threadIdx.x;
    const int b   = blockIdx.x;
    const float* xrow = x + (size_t)b * (3 * 32 * 32);
    for (int idx = tid; idx < 3 * 34 * 34; idx += 256) {
        const int c  = idx / 1156;
        const int rm = idx - c * 1156;
        const int yy = rm / 34, xx = rm - yy * 34;
        const int iy = yy - 1, ix = xx - 1;
        float v = 0.f;
        if ((unsigned)iy < 32u && (unsigned)ix < 32u) v = xrow[c * 1024 + iy * 32 + ix];
        xs[idx] = v;
    }
    for (int idx = tid; idx < 16 * 27; idx += 256) ws[idx] = w[idx];
    if (tid < 16) bb[tid] = bsrc[tid];
    __syncthreads();
    const int y = tid >> 4, xo = tid & 15;
    float acc[16];
#pragma unroll
    for (int co = 0; co < 16; ++co) acc[co] = bb[co];
#pragma unroll
    for (int ci = 0; ci < 3; ++ci)
#pragma unroll
        for (int dy = 0; dy < 3; ++dy)
#pragma unroll
            for (int dx = 0; dx < 3; ++dx) {
                const float v = xs[ci * 1156 + (2 * y + dy) * 34 + (2 * xo + dx)];
                const int wb = ci * 9 + dy * 3 + dx;
#pragma unroll
                for (int co = 0; co < 16; ++co) acc[co] += v * ws[co * 27 + wb];
            }
    float* orow = out + (size_t)b * DM;
#pragma unroll
    for (int co = 0; co < 16; ++co) orow[co * 256 + y * 16 + xo] = fmaxf(acc[co], 0.f);
}

// ---------------------------------------------------------------------------
// gate: logits = flat @ gate_w + gate_b ; write logits; route[b] = argmax.
// One wave per token (wave32), butterfly reduction.
// ---------------------------------------------------------------------------
__global__ __launch_bounds__(256)
void gate_kernel(const float* __restrict__ flat, const float* __restrict__ gw,
                 const float* __restrict__ gb, float* __restrict__ logits_out,
                 int* __restrict__ route) {
    const int lane = threadIdx.x & 31;
    const int wv   = threadIdx.x >> 5;
    const int b    = blockIdx.x * 8 + wv;
    const float* frow = flat + (size_t)b * DM;
    float a0 = 0.f, a1 = 0.f, a2 = 0.f, a3 = 0.f;
    for (int k = lane; k < DM; k += 32) {
        const float xv = frow[k];
        const float4 g = ((const float4*)gw)[k];
        a0 += xv * g.x; a1 += xv * g.y; a2 += xv * g.z; a3 += xv * g.w;
    }
#pragma unroll
    for (int off = 16; off > 0; off >>= 1) {
        a0 += __shfl_xor(a0, off, 32);
        a1 += __shfl_xor(a1, off, 32);
        a2 += __shfl_xor(a2, off, 32);
        a3 += __shfl_xor(a3, off, 32);
    }
    if (lane == 0) {
        a0 += gb[0]; a1 += gb[1]; a2 += gb[2]; a3 += gb[3];
        *(float4*)(logits_out + (size_t)b * 4) = make_float4(a0, a1, a2, a3);
        int best = 0; float bv = a0;
        if (a1 > bv) { bv = a1; best = 1; }
        if (a2 > bv) { bv = a2; best = 2; }
        if (a3 > bv) { bv = a3; best = 3; }
        route[b] = best;
    }
}

// ---------------------------------------------------------------------------
// selected-expert 3x3 conv (pad 1) + relu, output stored as f16 (GEMM A).
// ---------------------------------------------------------------------------
__global__ __launch_bounds__(256)
void expert_conv_kernel(const float* __restrict__ flat, const int* __restrict__ route,
                        const float* __restrict__ ew, const float* __restrict__ eb,
                        _Float16* __restrict__ out) {
    __shared__ float xs[16 * 18 * 18];
    __shared__ float ws[16 * 16 * 9];
    __shared__ float bs[16];
    const int tid = threadIdx.x;
    const int b   = blockIdx.x;
    const int e   = route[b];
    const float* frow = flat + (size_t)b * DM;
    for (int idx = tid; idx < 16 * 18 * 18; idx += 256) {
        const int c  = idx / 324;
        const int rm = idx - c * 324;
        const int yy = rm / 18, xx = rm - yy * 18;
        const int iy = yy - 1, ix = xx - 1;
        float v = 0.f;
        if ((unsigned)iy < 16u && (unsigned)ix < 16u) v = frow[c * 256 + iy * 16 + ix];
        xs[idx] = v;
    }
    for (int idx = tid; idx < 16 * 16 * 9; idx += 256) ws[idx] = ew[e * 2304 + idx];
    if (tid < 16) bs[tid] = eb[e * 16 + tid];
    __syncthreads();
    const int y = tid >> 4, xo = tid & 15;
    float acc[16];
#pragma unroll
    for (int co = 0; co < 16; ++co) acc[co] = bs[co];
    for (int ci = 0; ci < 16; ++ci)
#pragma unroll
        for (int dy = 0; dy < 3; ++dy)
#pragma unroll
            for (int dx = 0; dx < 3; ++dx) {
                const float v = xs[ci * 324 + (y + dy) * 18 + (xo + dx)];
                const int wb = ci * 9 + dy * 3 + dx;
#pragma unroll
                for (int co = 0; co < 16; ++co) acc[co] += v * ws[co * 144 + wb];
            }
    _Float16* orow = out + (size_t)b * DM;
#pragma unroll
    for (int co = 0; co < 16; ++co)
        orow[co * 256 + y * 16 + xo] = (_Float16)fmaxf(acc[co], 0.f);
}

// ---------------------------------------------------------------------------
// WMMA GEMM: C[8192,4096] = A(f16) x Bt(f16,[N][K])^T + bias -> fp32.
// Block tile 128x64, 8 waves, wave tile 16x64. K-step 64, double-buffered
// LDS, one barrier per step. All 20 ds_loads of a stage are hoisted into
// 10 simultaneously-live fragments so the 8 WMMAs issue with partial
// dscnt waits instead of full stalls.
// ---------------------------------------------------------------------------
__global__ __launch_bounds__(256)
void gemm_wmma_f16_kernel(const _Float16* __restrict__ A, const _Float16* __restrict__ Bt,
                          const float* __restrict__ bias, float* __restrict__ C) {
    constexpr int KS  = 64;                 // K per stage
    constexpr int NS  = DM / KS;            // 64 stages
    constexpr int LDA = 72;                 // halves/row: 144B (16B-aligned), 36-bank stride
    __shared__ _Float16 As[2][128 * LDA];
    __shared__ _Float16 Bs[2][64 * LDA];

    const int tid   = threadIdx.x;
    const int lane  = tid & 31;
    const int wv    = tid >> 5;             // 0..7
    const int mbase = blockIdx.y * 128;
    const int nbase = blockIdx.x * 64;
    const int l15   = lane & 15;
    const int hi    = lane >> 4;            // 0/1
    const int kbA   = hi * 8;               // A frag: K 0-7/16-23 vs 8-15/24-31
    const int kbB   = hi * 8;               // B frag: K 0-7/16-23 vs 8-15/24-31

    v8f acc[4] = {};

    // staging: A = 128 rows x 64 halves (32 halves/thread), B = 64 x 64 (16/thread)
    const int arow = tid >> 1, aseg = tid & 1;
    const int brow = tid & 63, bseg = tid >> 6;
    const _Float16* gA = A  + (size_t)(mbase + arow) * DM + aseg * 32;
    const _Float16* gB = Bt + (size_t)(nbase + brow) * DM + bseg * 16;
    const int sAoff = arow * LDA + aseg * 32;
    const int sBoff = brow * LDA + bseg * 16;

    {   // prologue: stage tile 0 into buffer 0
        const v8h* pa = (const v8h*)gA;
        const v8h* pb = (const v8h*)gB;
        *(v8h*)&As[0][sAoff +  0] = pa[0];
        *(v8h*)&As[0][sAoff +  8] = pa[1];
        *(v8h*)&As[0][sAoff + 16] = pa[2];
        *(v8h*)&As[0][sAoff + 24] = pa[3];
        *(v8h*)&Bs[0][sBoff + 0]  = pb[0];
        *(v8h*)&Bs[0][sBoff + 8]  = pb[1];
    }

    for (int s = 0; s < NS; ++s) {
        __syncthreads();                    // buf[s&1] fully staged; prev reads done
        const int cur = s & 1, nxt = cur ^ 1;

        v8h ra0 = {}, ra1 = {}, ra2 = {}, ra3 = {}, rb0 = {}, rb1 = {};
        const bool more = (s + 1 < NS);
        if (more) {                         // issue next-stage global loads now
            const v8h* pa = (const v8h*)(gA + (s + 1) * KS);
            ra0 = pa[0]; ra1 = pa[1]; ra2 = pa[2]; ra3 = pa[3];
            const v8h* pb = (const v8h*)(gB + (s + 1) * KS);
            rb0 = pb[0]; rb1 = pb[1];
            if (s + 2 < NS) {               // global_prefetch_b8 one more stage out
                __builtin_prefetch(gA + (s + 2) * KS, 0, 1);
                __builtin_prefetch(gB + (s + 2) * KS, 0, 1);
            }
        }

        // ---- hoist ALL fragment loads for this stage (10 live v16h) ----
        v16h afr[2];
        v16h bfr[2][4];
#pragma unroll
        for (int kh = 0; kh < 2; ++kh) {
            const _Float16* ap = &As[cur][(wv * 16 + l15) * LDA + kh * 32];
            const v8h a0 = *(const v8h*)(ap + kbA);
            const v8h a1 = *(const v8h*)(ap + kbA + 16);
#pragma unroll
            for (int j = 0; j < 8; ++j) { afr[kh][j] = a0[j]; afr[kh][8 + j] = a1[j]; }
#pragma unroll
            for (int t = 0; t < 4; ++t) {
                const _Float16* bp = &Bs[cur][(t * 16 + l15) * LDA + kh * 32];
                const v8h b0 = *(const v8h*)(bp + kbB);
                const v8h b1 = *(const v8h*)(bp + kbB + 16);
#pragma unroll
                for (int j = 0; j < 8; ++j) { bfr[kh][t][j] = b0[j]; bfr[kh][t][8 + j] = b1[j]; }
            }
        }
        // ---- 8 WMMAs back-to-back ----
#pragma unroll
        for (int kh = 0; kh < 2; ++kh)
#pragma unroll
            for (int t = 0; t < 4; ++t)
                acc[t] = __builtin_amdgcn_wmma_f32_16x16x32_f16(
                    false, afr[kh], false, bfr[kh][t], (short)0, acc[t], false, false);

        if (more) {                         // stage next buffer (post-compute)
            *(v8h*)&As[nxt][sAoff +  0] = ra0;
            *(v8h*)&As[nxt][sAoff +  8] = ra1;
            *(v8h*)&As[nxt][sAoff + 16] = ra2;
            *(v8h*)&As[nxt][sAoff + 24] = ra3;
            *(v8h*)&Bs[nxt][sBoff + 0]  = rb0;
            *(v8h*)&Bs[nxt][sBoff + 8]  = rb1;
        }
    }

    const int rbase = mbase + wv * 16 + hi * 8;   // C/D layout: vgpr r -> M = r + 8*hi
#pragma unroll
    for (int t = 0; t < 4; ++t) {
        const int col = nbase + t * 16 + l15;
        const float bv = bias[col];
#pragma unroll
        for (int r = 0; r < 8; ++r)
            C[(size_t)(rbase + r) * DM + col] = acc[t][r] + bv;
    }
}

// ---------------------------------------------------------------------------
// head: conv(8ch,3x3,s2,p1)+relu -> maxpool2 -> fc1+relu -> fc2
// ---------------------------------------------------------------------------
__global__ __launch_bounds__(128)
void head_kernel(const float* __restrict__ flat, const float* __restrict__ lw,
                 const float* __restrict__ lb, const float* __restrict__ w1,
                 const float* __restrict__ b1, const float* __restrict__ w2,
                 const float* __restrict__ b2, float* __restrict__ out) {
    __shared__ float hs[DM];
    __shared__ float wsv[8 * 16 * 9];
    __shared__ float c2[512];
    __shared__ float pl[128];
    __shared__ float f1[16];
    const int tid = threadIdx.x;
    const int b   = blockIdx.x;
    const float* frow = flat + (size_t)b * DM;
    for (int idx = tid; idx < DM; idx += 128) hs[idx] = frow[idx];
    for (int idx = tid; idx < 8 * 16 * 9; idx += 128) wsv[idx] = lw[idx];
    __syncthreads();
#pragma unroll
    for (int i = 0; i < 4; ++i) {
        const int o  = tid + i * 128;             // o = co*64 + y*8 + x
        const int co = o >> 6;
        const int rm = o & 63;
        const int y  = rm >> 3, xo = rm & 7;
        float acc = lb[co];
        for (int ci = 0; ci < 16; ++ci)
#pragma unroll
            for (int dy = 0; dy < 3; ++dy) {
                const int iy = 2 * y + dy - 1;
                if ((unsigned)iy >= 16u) continue;
#pragma unroll
                for (int dx = 0; dx < 3; ++dx) {
                    const int ix = 2 * xo + dx - 1;
                    if ((unsigned)ix >= 16u) continue;
                    acc += hs[ci * 256 + iy * 16 + ix] * wsv[co * 144 + ci * 9 + dy * 3 + dx];
                }
            }
        c2[o] = fmaxf(acc, 0.f);
    }
    __syncthreads();
    {
        const int co = tid >> 4, y = (tid >> 2) & 3, xo = tid & 3;
        float m = c2[co * 64 + (2 * y) * 8 + 2 * xo];
        m = fmaxf(m, c2[co * 64 + (2 * y) * 8 + 2 * xo + 1]);
        m = fmaxf(m, c2[co * 64 + (2 * y + 1) * 8 + 2 * xo]);
        m = fmaxf(m, c2[co * 64 + (2 * y + 1) * 8 + 2 * xo + 1]);
        pl[tid] = m;                              // co*16 + y*4 + x == tid
    }
    __syncthreads();
    if (tid < 16) {
        float acc = b1[tid];
        for (int i = 0; i < 128; ++i) acc += pl[i] * w1[i * 16 + tid];
        f1[tid] = fmaxf(acc, 0.f);
    }
    __syncthreads();
    if (tid < 10) {
        float acc = b2[tid];
#pragma unroll
        for (int i = 0; i < 16; ++i) acc += f1[i] * w2[i * 10 + tid];
        out[(size_t)b * 10 + tid] = acc;
    }
}

// ---------------------------------------------------------------------------
extern "C" void kernel_launch(void* const* d_in, const int* in_sizes, int n_in,
                              void* d_out, int out_size, void* d_ws, size_t ws_size,
                              hipStream_t stream) {
    const float* x        = (const float*)d_in[0];
    const float* conv1_w  = (const float*)d_in[1];
    const float* conv1_b  = (const float*)d_in[2];
    const float* expert_w = (const float*)d_in[3];
    const float* expert_b = (const float*)d_in[4];
    const float* gate_w   = (const float*)d_in[5];
    const float* gate_b   = (const float*)d_in[6];
    const float* trans_w  = (const float*)d_in[7];
    const float* trans_b  = (const float*)d_in[8];
    const float* last_w   = (const float*)d_in[9];
    const float* last_b   = (const float*)d_in[10];
    const float* fc1_w    = (const float*)d_in[11];
    const float* fc1_b    = (const float*)d_in[12];
    const float* fc2_w    = (const float*)d_in[13];
    const float* fc2_b    = (const float*)d_in[14];
    float* out = (float*)d_out;

    char* ws = (char*)d_ws;
    _Float16* Wt   = (_Float16*)ws;  ws += (size_t)DM * DM * sizeof(_Float16);
    float* bufA    = (float*)ws;     ws += (size_t)BTOK * DM * sizeof(float);
    _Float16* bufM = (_Float16*)ws;  ws += (size_t)BTOK * DM * sizeof(_Float16);
    int* route     = (int*)ws;

    transpose_convert_kernel<<<dim3(DM / 32, DM / 32), 256, 0, stream>>>(trans_w, Wt);
    conv1_kernel<<<BTOK, 256, 0, stream>>>(x, conv1_w, conv1_b, bufA);

    for (int iter = 0; iter < 2; ++iter) {
        float* logits_out = out + (size_t)BTOK * 10 + (size_t)iter * BTOK * 4;
        gate_kernel<<<BTOK / 8, 256, 0, stream>>>(bufA, gate_w, gate_b, logits_out, route);
        expert_conv_kernel<<<BTOK, 256, 0, stream>>>(bufA, route, expert_w, expert_b, bufM);
        gemm_wmma_f16_kernel<<<dim3(DM / 64, BTOK / 128), 256, 0, stream>>>(bufM, Wt, trans_b, bufA);
    }

    head_kernel<<<BTOK, 128, 0, stream>>>(bufA, last_w, last_b, fc1_w, fc1_b, fc2_w, fc2_b, out);
}
